// MessagePassing_3135326126152
// MI455X (gfx1250) — compile-verified, matrix operand order checked by reading, not run on hardware
//
#include <hip/hip_runtime.h>
#include <hip/hip_bf16.h>

typedef __attribute__((ext_vector_type(16))) __bf16 v16bf;
typedef __attribute__((ext_vector_type(8)))  __bf16 v8bf;
typedef __attribute__((ext_vector_type(8)))  float  v8f;
typedef __attribute__((ext_vector_type(4)))  float  v4f;

#define N_NODES   50000
#define N_EDGES   800000
#define NODE_DIM  128
#define RAD_DIM   64
#define ANG_DIM   64
#define COMB      128
#define IN_DIM    384
#define MSG_DIM   256          // NODE_DIM + COMB
#define TILE      128          // edges (or nodes) per workgroup
#define NTHREADS  256          // 8 waves of 32

// LDS strides (bf16 elements), multiples of 8 so 16B vector accesses stay
// aligned, padded to perturb bank mapping.
#define LD0 392                // buffer 0, max width 384
#define LD1 200                // buffer 1, max width 192
#define LDN 264                // node-kernel A buffer, width 256
#define EDGE_SMEM ((TILE * LD0 + TILE * LD1) * 2)   // ~148 KB -> 2 WGs/WGP
#define NODE_SMEM (TILE * LDN * 2)                  // ~66 KB

struct EdgeWeightsPacked {
  const __bf16* W[6];   // packed bf16, [N][Kpad], zero-padded K
  const float*  b[6];
};

// One MLP layer on a 128-row tile in LDS (bf16) -> LDS (bf16).
// D = act(A @ Wp^T + b). Each wave owns 16 output columns; the B fragment
// (two 16B global loads from the packed bf16 weights) is loaded once per
// k-step and reused across 8 row tiles (A fragments: two ds_load_b128 each).
__device__ __forceinline__ void gemm_layer_bf16(
    const __bf16* __restrict__ Wp, const float* __restrict__ bias,
    const __bf16* in, int ldin, __bf16* out, int ldout,
    int Kpad, int N, bool relu, int tid)
{
  const int wave = tid >> 5;
  const int lane = tid & 31;
  const int l16  = lane & 15;
  const int kSel = (lane < 16) ? 0 : 8;      // ISA 16-bit A/B lane K base
  const int mOff = (lane >> 4) << 3;         // C/D: lanes 16-31 hold rows +8
  const int nCT  = N >> 4;
  for (int ct = wave; ct < nCT; ct += 8) {
    const int n = (ct << 4) + l16;           // this lane's output column
    v8f acc[8];
#pragma unroll
    for (int r = 0; r < 8; ++r)
#pragma unroll
      for (int i = 0; i < 8; ++i) acc[r][i] = 0.0f;

    for (int k0 = 0; k0 < Kpad; k0 += 32) {
      // B fragment: column n of B == row n of packed W, two 8-elem K runs.
      const v8bf* bp = (const v8bf*)(Wp + n * Kpad + k0 + kSel);
      const v8bf blo = bp[0];                // K = kSel .. kSel+7
      const v8bf bhi = bp[2];                // K = kSel+16 .. kSel+23
      v16bf bfrag;
#pragma unroll
      for (int i = 0; i < 8; ++i) { bfrag[i] = blo[i]; bfrag[8 + i] = bhi[i]; }
#pragma unroll
      for (int rt = 0; rt < 8; ++rt) {
        const v8bf* ap = (const v8bf*)(in + (rt * 16 + l16) * ldin + k0 + kSel);
        const v8bf alo = ap[0];
        const v8bf ahi = ap[2];
        v16bf afrag;
#pragma unroll
        for (int i = 0; i < 8; ++i) { afrag[i] = alo[i]; afrag[8 + i] = ahi[i]; }
        acc[rt] = __builtin_amdgcn_wmma_f32_16x16x32_bf16(
            false, afrag, false, bfrag, (short)0, acc[rt], false, false);
      }
    }
    // Epilogue: bias + optional ReLU, store bf16 back to LDS.
    const float bv = bias[n];
#pragma unroll
    for (int rt = 0; rt < 8; ++rt)
#pragma unroll
      for (int r = 0; r < 8; ++r) {
        float v = acc[rt][r] + bv;
        if (relu) v = fmaxf(v, 0.0f);
        out[(rt * 16 + r + mOff) * ldout + n] = (__bf16)v;
      }
  }
}

// Pack one f32 [N][Kact] weight matrix into bf16 [N][Kpad] (zero-pad K).
__global__ void pack_weights_kernel(const float* __restrict__ W,
                                    __bf16* __restrict__ out,
                                    int N, int Kact, int Kpad) {
  const int idx = blockIdx.x * blockDim.x + threadIdx.x;
  if (idx >= N * Kpad) return;
  const int n = idx / Kpad, k = idx - n * Kpad;
  out[idx] = (__bf16)((k < Kact) ? W[n * Kact + k] : 0.0f);
}

__global__ __launch_bounds__(NTHREADS) void edge_mlp_kernel(
    const float* __restrict__ nodef, const float* __restrict__ rad,
    const float* __restrict__ ang, const int* __restrict__ src,
    const int* __restrict__ dst, EdgeWeightsPacked ew,
    float* __restrict__ out_edge)
{
  extern __shared__ __align__(16) __bf16 smem[];
  __bf16* buf0 = smem;
  __bf16* buf1 = smem + TILE * LD0;
  const int tid = threadIdx.x;
  const long e0 = (long)blockIdx.x * TILE;

  // Prefetch next tile's dense radial/angular rows (global_prefetch_b8).
  if (e0 + TILE < N_EDGES) {
    const long en = e0 + TILE;
    __builtin_prefetch((const char*)(rad + en * RAD_DIM) + tid * 128, 0, 1);
    __builtin_prefetch((const char*)(ang + en * ANG_DIM) + tid * 128, 0, 1);
  }

  // Phase 1: gather 128 edges x 384 inputs into LDS as bf16 (8-wide vectors;
  // all segment boundaries are multiples of 8 so a chunk never straddles).
  for (int idx = tid; idx < TILE * (IN_DIM / 8); idx += NTHREADS) {
    const int e = idx / (IN_DIM / 8);
    const int c = (idx - e * (IN_DIM / 8)) * 8;
    const long ge = e0 + e;
    const float* sp;
    if (c < NODE_DIM)                 sp = nodef + (long)src[ge] * NODE_DIM + c;
    else if (c < 2 * NODE_DIM)        sp = nodef + (long)dst[ge] * NODE_DIM + (c - NODE_DIM);
    else if (c < 2 * NODE_DIM + RAD_DIM) sp = rad + ge * RAD_DIM + (c - 2 * NODE_DIM);
    else                              sp = ang + ge * ANG_DIM + (c - 2 * NODE_DIM - RAD_DIM);
    const v4f a = ((const v4f*)sp)[0];
    const v4f b = ((const v4f*)sp)[1];
    v8bf o;
#pragma unroll
    for (int i = 0; i < 4; ++i) { o[i] = (__bf16)a[i]; o[4 + i] = (__bf16)b[i]; }
    *((v8bf*)(buf0 + e * LD0 + c)) = o;
  }
  __syncthreads();

  gemm_layer_bf16(ew.W[0], ew.b[0], buf0, LD0, buf1, LD1, 384, 192, true, tid);
  __syncthreads();
  gemm_layer_bf16(ew.W[1], ew.b[1], buf1, LD1, buf0, LD0, 192,  96, true, tid);
  __syncthreads();
  gemm_layer_bf16(ew.W[2], ew.b[2], buf0, LD0, buf1, LD1,  96,  48, true, tid);
  // Zero-pad activation cols 48..63 so the next layer can run K=64 steps.
  for (int idx = tid; idx < TILE * 2; idx += NTHREADS) {
    v8bf z;
#pragma unroll
    for (int i = 0; i < 8; ++i) z[i] = (__bf16)0.0f;
    *((v8bf*)(buf1 + (idx >> 1) * LD1 + 48 + (idx & 1) * 8)) = z;
  }
  __syncthreads();
  gemm_layer_bf16(ew.W[3], ew.b[3], buf1, LD1, buf0, LD0,  64,  32, true, tid);
  __syncthreads();
  gemm_layer_bf16(ew.W[4], ew.b[4], buf0, LD0, buf1, LD1,  32,  64, true, tid);
  __syncthreads();
  gemm_layer_bf16(ew.W[5], ew.b[5], buf1, LD1, buf0, LD0,  64, 128, false, tid);
  __syncthreads();

  // Phase 3: write final 128-wide edge features as f32 (vectorized).
  for (int idx = tid; idx < TILE * (COMB / 8); idx += NTHREADS) {
    const int e = idx >> 4, c = (idx & 15) * 8;
    const v8bf o = *((const v8bf*)(buf0 + e * LD0 + c));
    v4f f0, f1;
#pragma unroll
    for (int i = 0; i < 4; ++i) { f0[i] = (float)o[i]; f1[i] = (float)o[4 + i]; }
    float* q = out_edge + (e0 + e) * COMB + c;
    ((v4f*)q)[0] = f0;
    ((v4f*)q)[1] = f1;
  }
}

__global__ void zero_kernel(float* __restrict__ p, long n) {
  const long i = (long)blockIdx.x * blockDim.x + threadIdx.x;
  if (i < n) p[i] = 0.0f;
}

__global__ void degree_kernel(const int* __restrict__ dst, float* __restrict__ deg) {
  const int e = blockIdx.x * blockDim.x + threadIdx.x;
  if (e < N_EDGES) atomicAdd(&deg[dst[e]], 1.0f);
}

// One block per edge; 256 threads cover the 256 message columns (segment_sum).
__global__ __launch_bounds__(NTHREADS) void scatter_kernel(
    const float* __restrict__ nodef, const float* __restrict__ rad,
    const float* __restrict__ ang, const int* __restrict__ src,
    const int* __restrict__ dst, const float* __restrict__ deg,
    float* __restrict__ msg)
{
  const int e = blockIdx.x;
  const int c = threadIdx.x;
  const int s = src[e];
  const int d = dst[e];
  const float scale = 1.0f / deg[s];   // in-degree of SOURCE node (faithful)
  float v;
  if (c < NODE_DIM)                v = nodef[(long)src[s] * NODE_DIM + c]; // src[src[e]] (faithful bug)
  else if (c < NODE_DIM + RAD_DIM) v = rad[(long)e * RAD_DIM + (c - NODE_DIM)];
  else                             v = ang[(long)e * ANG_DIM + (c - NODE_DIM - RAD_DIM)];
  atomicAdd(&msg[(long)d * MSG_DIM + c], v * scale);
}

// Node finalize + 50000x256x128 WMMA GEMM + SiLU.
__global__ __launch_bounds__(NTHREADS) void node_update_kernel(
    const float* __restrict__ nodef, const float* __restrict__ msg,
    const float* __restrict__ deg, const __bf16* __restrict__ Wp,
    const float* __restrict__ bn, float* __restrict__ out_node)
{
  extern __shared__ __align__(16) __bf16 smem[];
  __bf16* bufA = smem;
  const int tid  = threadIdx.x;
  const int base = blockIdx.x * TILE;

  // Build A = ((msg + [nf/sqrt(deg), 0]) / sqrt(deg)) as bf16 in LDS.
  for (int idx = tid; idx < TILE * (MSG_DIM / 8); idx += NTHREADS) {
    const int r = idx >> 5;
    const int c = (idx & 31) * 8;
    const int node = base + r;
    v8bf o;
    if (node < N_NODES) {
      const float rs = rsqrtf(deg[node]);
      const v4f m0 = ((const v4f*)(msg + (long)node * MSG_DIM + c))[0];
      const v4f m1 = ((const v4f*)(msg + (long)node * MSG_DIM + c))[1];
      v4f f0 = m0, f1 = m1;
      if (c < NODE_DIM) {
        const v4f n0 = ((const v4f*)(nodef + (long)node * NODE_DIM + c))[0];
        const v4f n1 = ((const v4f*)(nodef + (long)node * NODE_DIM + c))[1];
#pragma unroll
        for (int i = 0; i < 4; ++i) { f0[i] += n0[i] * rs; f1[i] += n1[i] * rs; }
      }
#pragma unroll
      for (int i = 0; i < 4; ++i) { o[i] = (__bf16)(f0[i] * rs); o[4 + i] = (__bf16)(f1[i] * rs); }
    } else {
#pragma unroll
      for (int i = 0; i < 8; ++i) o[i] = (__bf16)0.0f;
    }
    *((v8bf*)(bufA + r * LDN + c)) = o;
  }
  __syncthreads();

  const int wave = tid >> 5, lane = tid & 31, l16 = lane & 15;
  const int kSel = (lane < 16) ? 0 : 8;
  const int mOff = (lane >> 4) << 3;
  const int n = (wave << 4) + l16;           // 8 waves x 16 cols = 128 outputs
  v8f acc[8];
#pragma unroll
  for (int r = 0; r < 8; ++r)
#pragma unroll
    for (int i = 0; i < 8; ++i) acc[r][i] = 0.0f;

  for (int k0 = 0; k0 < MSG_DIM; k0 += 32) {
    const v8bf* bp = (const v8bf*)(Wp + n * MSG_DIM + k0 + kSel);
    const v8bf blo = bp[0];
    const v8bf bhi = bp[2];
    v16bf bfrag;
#pragma unroll
    for (int i = 0; i < 8; ++i) { bfrag[i] = blo[i]; bfrag[8 + i] = bhi[i]; }
#pragma unroll
    for (int rt = 0; rt < 8; ++rt) {
      const v8bf* ap = (const v8bf*)(bufA + (rt * 16 + l16) * LDN + k0 + kSel);
      const v8bf alo = ap[0];
      const v8bf ahi = ap[2];
      v16bf afrag;
#pragma unroll
      for (int i = 0; i < 8; ++i) { afrag[i] = alo[i]; afrag[8 + i] = ahi[i]; }
      acc[rt] = __builtin_amdgcn_wmma_f32_16x16x32_bf16(
          false, afrag, false, bfrag, (short)0, acc[rt], false, false);
    }
  }
  const float bv = bn[n];
#pragma unroll
  for (int rt = 0; rt < 8; ++rt)
#pragma unroll
    for (int r = 0; r < 8; ++r) {
      const int node = base + rt * 16 + r + mOff;
      if (node < N_NODES) {
        float v = acc[rt][r] + bv;
        v = v / (1.0f + __expf(-v));         // SiLU
        out_node[(long)node * NODE_DIM + n] = v;
      }
    }
}

extern "C" void kernel_launch(void* const* d_in, const int* in_sizes, int n_in,
                              void* d_out, int out_size, void* d_ws, size_t ws_size,
                              hipStream_t stream)
{
  const float* nodef = (const float*)d_in[0];
  const float* rad   = (const float*)d_in[1];
  const float* ang   = (const float*)d_in[2];
  const int*   ei    = (const int*)d_in[3];        // [2, E] int32
  const int*   src   = ei;
  const int*   dst   = ei + N_EDGES;

  const float* Wsrc[7];
  const float* bsrc[7];
  for (int i = 0; i < 6; ++i) {
    Wsrc[i] = (const float*)d_in[4 + 2 * i];
    bsrc[i] = (const float*)d_in[5 + 2 * i];
  }
  Wsrc[6] = (const float*)d_in[16];   // Wn [128, 256]
  bsrc[6] = (const float*)d_in[17];   // bn [128]

  float* out_node = (float*)d_out;                          // [N, 128]
  float* out_edge = out_node + (size_t)N_NODES * NODE_DIM;  // [E, 128]

  // Workspace layout: deg [N] f32 | msg [N,256] f32 | packed bf16 weights
  float* deg = (float*)d_ws;
  float* msg = deg + N_NODES;
  __bf16* wbase = (__bf16*)((char*)d_ws + (size_t)N_NODES * (1 + MSG_DIM) * 4);

  static const int KACT[7] = {384, 192, 96, 48, 32, 64, 256};
  static const int KPAD[7] = {384, 192, 96, 64, 32, 64, 256};
  static const int NOUT[7] = {192,  96, 48, 32, 64, 128, 128};

  __bf16* wp[7];
  size_t off = 0;
  for (int i = 0; i < 7; ++i) { wp[i] = wbase + off; off += (size_t)NOUT[i] * KPAD[i]; }

  // Pack all weight matrices to bf16 with K zero-padded to a multiple of 32.
  for (int i = 0; i < 7; ++i) {
    const int total = NOUT[i] * KPAD[i];
    pack_weights_kernel<<<(total + 255) / 256, 256, 0, stream>>>(
        Wsrc[i], wp[i], NOUT[i], KACT[i], KPAD[i]);
  }

  EdgeWeightsPacked ew;
  for (int i = 0; i < 6; ++i) { ew.W[i] = wp[i]; ew.b[i] = bsrc[i]; }

  const long nz = (long)N_NODES * (1 + MSG_DIM);
  zero_kernel<<<(unsigned)((nz + 255) / 256), 256, 0, stream>>>(deg, nz);
  degree_kernel<<<(N_EDGES + 255) / 256, 256, 0, stream>>>(dst, deg);
  scatter_kernel<<<N_EDGES, NTHREADS, 0, stream>>>(nodef, rad, ang, src, dst, deg, msg);
  edge_mlp_kernel<<<N_EDGES / TILE, NTHREADS, EDGE_SMEM, stream>>>(
      nodef, rad, ang, src, dst, ew, out_edge);
  node_update_kernel<<<(N_NODES + TILE - 1) / TILE, NTHREADS, NODE_SMEM, stream>>>(
      nodef, msg, deg, wp[6], bsrc[6], out_node);
}
